// MCTCSelfAttention_68418829025822
// MI455X (gfx1250) — matrix operand order repaired and checked by reading, not compiled, not run on hardware
//
#include <hip/hip_runtime.h>

// ---------------------------------------------------------------------------
// MCTC self-attention for MI455X (gfx1250, wave32, WMMA f16 -> f32)
//   B=4, S=920, D=1536, H=4, Dh=384, rel bias: scores[q,k] += q[q]·E[919+k-q]
// ---------------------------------------------------------------------------

typedef _Float16 h4  __attribute__((ext_vector_type(4)));
typedef _Float16 h8  __attribute__((ext_vector_type(8)));
typedef _Float16 h16 __attribute__((ext_vector_type(16)));
typedef float    f8  __attribute__((ext_vector_type(8)));

#define S_LEN   920
#define DH      384
#define DMODEL  1536
#define MROWS   3680          // B*S
#define LREL    1839          // 2*S-1
#define LPAD    1856          // padded rel rows (gather reads l up to 1846)
#define SPAD    928           // padded S pitch
#define INV_SQRT_DH 0.05103103630798288f
#define NEG_INF (-__builtin_huge_valf())

// A-fragment, 16-bit 16x32 (M x K): lane m=L&15; halves 0..7 = K(koff..koff+7),
// halves 8..15 = K(16+koff..), koff = 8*(L>>4). Caller bakes koff into p.
static __device__ inline h16 fragA(const _Float16* p) {
  h8 lo = *(const h8*)p;
  h8 hi = *(const h8*)(p + 16);
  return __builtin_shufflevector(lo, hi, 0,1,2,3,4,5,6,7,8,9,10,11,12,13,14,15);
}
// B-fragment, 16-bit 32x16 (K x N): lane n=L&15; halves 0..7 = K(kb..kb+7),
// halves 8..15 = K(16+kb..), kb = 8*(L>>4). Caller bakes kb into p; column must be contiguous in K.
static __device__ inline h16 fragB(const _Float16* p) {
  h8 lo = *(const h8*)p;
  h8 hi = *(const h8*)(p + 16);
  return __builtin_shufflevector(lo, hi, 0,1,2,3,4,5,6,7,8,9,10,11,12,13,14,15);
}
static __device__ inline f8 wmma16(h16 a, h16 b, f8 c) {
  return __builtin_amdgcn_wmma_f32_16x16x32_f16(false, a, false, b, (short)0, c,
                                                false, false);
}
static __device__ inline f8 zero8() {
  f8 z = {0.f,0.f,0.f,0.f,0.f,0.f,0.f,0.f};
  return z;
}

// ---------------------------------------------------------------------------
// Kernel 1: QKV projection. C = hidden(3680x1536) * W(1536x1536), f32 in,
// f16 out. sel=0 -> q (scaled by 1/sqrt(Dh)) [bh][s][d]; sel=1 -> k same;
// sel=2 -> v stored TRANSPOSED [bh][d][s_pad928].
// Block: 256 thr = 8 waves; block tile 128x128; wave strip 16x128.
// ---------------------------------------------------------------------------
__global__ __launch_bounds__(256)
void mctc_qkv_kernel(const float* __restrict__ hid,
                     const float* __restrict__ qw,
                     const float* __restrict__ kw,
                     const float* __restrict__ vw,
                     _Float16* __restrict__ qout,
                     _Float16* __restrict__ kout,
                     _Float16* __restrict__ vtout) {
  __shared__ _Float16 aT[128 * 32];   // [row][k]
  __shared__ _Float16 bT[128 * 32];   // [n][k]  (W transposed)

  const int tid  = threadIdx.x;
  const int wave = tid >> 5, lane = tid & 31;
  const int m0   = blockIdx.x * 128;
  const int n0   = blockIdx.y * 128;
  const int sel  = blockIdx.z;
  const float* W = (sel == 0) ? qw : ((sel == 1) ? kw : vw);

  const int l15   = lane & 15;
  const int hsel  = lane >> 4;
  const int koffA = hsel * 8;
  const int koffB = hsel * 8;

  f8 acc[8];
  #pragma unroll
  for (int t = 0; t < 8; ++t) acc[t] = zero8();

  for (int k0 = 0; k0 < DMODEL; k0 += 32) {
    __syncthreads();
    // stage A tile (128x32 f32 -> f16)
    #pragma unroll
    for (int i = 0; i < 4; ++i) {
      int f  = tid + 256 * i;
      int r  = f >> 3, kv = f & 7;
      int gr = m0 + r; if (gr > MROWS - 1) gr = MROWS - 1;
      const float4 v = *(const float4*)(hid + (size_t)gr * DMODEL + k0 + kv * 4);
      h4 hv = { (_Float16)v.x, (_Float16)v.y, (_Float16)v.z, (_Float16)v.w };
      *(h4*)&aT[r * 32 + kv * 4] = hv;
    }
    // stage W tile transposed (32 k-rows x 128 n-cols)
    #pragma unroll
    for (int i = 0; i < 4; ++i) {
      int f = tid + 256 * i;
      int r = f >> 5, nv = f & 31;
      const float4 v = *(const float4*)(W + (size_t)(k0 + r) * DMODEL + n0 + nv * 4);
      bT[(nv * 4 + 0) * 32 + r] = (_Float16)v.x;
      bT[(nv * 4 + 1) * 32 + r] = (_Float16)v.y;
      bT[(nv * 4 + 2) * 32 + r] = (_Float16)v.z;
      bT[(nv * 4 + 3) * 32 + r] = (_Float16)v.w;
    }
    __syncthreads();

    h16 a = fragA(&aT[(16 * wave + l15) * 32 + koffA]);
    #pragma unroll
    for (int t = 0; t < 8; ++t) {
      h16 b = fragB(&bT[(16 * t + l15) * 32 + koffB]);
      acc[t] = wmma16(a, b, acc[t]);
    }
  }

  const float scale = (sel == 0) ? INV_SQRT_DH : 1.0f;
  #pragma unroll
  for (int t = 0; t < 8; ++t) {
    #pragma unroll
    for (int r = 0; r < 8; ++r) {
      int mg = m0 + 16 * wave + r + koffA;     // C layout: m = r + 8*hsel
      int ng = n0 + 16 * t + l15;              //           n = lane&15
      if (mg < MROWS) {
        int b = mg / S_LEN, s = mg % S_LEN;
        int h = ng / DH,    d = ng % DH;
        int bh = b * 4 + h;
        _Float16 val = (_Float16)(acc[t][r] * scale);
        if (sel == 2) vtout[((size_t)bh * DH + d) * SPAD + s] = val;
        else if (sel == 0) qout[((size_t)bh * S_LEN + s) * DH + d] = val;
        else               kout[((size_t)bh * S_LEN + s) * DH + d] = val;
      }
    }
  }
}

// ---------------------------------------------------------------------------
// Kernel 2: rel[bh][l][s] = sum_d dist_emb[l][d] * q[bh][s][d]  (f16 out)
// GEMM (1839x384) x (384x920) per bh. Same 128x128 tiling.
// ---------------------------------------------------------------------------
__global__ __launch_bounds__(256)
void mctc_rel_kernel(const float* __restrict__ emb,
                     const _Float16* __restrict__ qws,
                     _Float16* __restrict__ relws) {
  __shared__ _Float16 aT[128 * 32];   // dist_emb rows
  __shared__ _Float16 bT[128 * 32];   // [s][d] = q rows (already B-layout)

  const int tid  = threadIdx.x;
  const int wave = tid >> 5, lane = tid & 31;
  const int m0 = blockIdx.x * 128;          // l
  const int n0 = blockIdx.y * 128;          // s
  const int bh = blockIdx.z;

  const int l15 = lane & 15, hsel = lane >> 4;
  const int koffA = hsel * 8, koffB = hsel * 8;
  const _Float16* qb = qws + (size_t)bh * S_LEN * DH;

  f8 acc[8];
  #pragma unroll
  for (int t = 0; t < 8; ++t) acc[t] = zero8();

  for (int k0 = 0; k0 < DH; k0 += 32) {
    __syncthreads();
    #pragma unroll
    for (int i = 0; i < 4; ++i) {
      int f  = tid + 256 * i;
      int r  = f >> 3, kv = f & 7;
      int gl = m0 + r; if (gl > LREL - 1) gl = LREL - 1;
      const float4 v = *(const float4*)(emb + (size_t)gl * DH + k0 + kv * 4);
      h4 hv = { (_Float16)v.x, (_Float16)v.y, (_Float16)v.z, (_Float16)v.w };
      *(h4*)&aT[r * 32 + kv * 4] = hv;
    }
    #pragma unroll
    for (int i = 0; i < 2; ++i) {
      int f = tid + 256 * i;
      int n = f >> 2, kv = f & 3;
      int gs = n0 + n; if (gs > S_LEN - 1) gs = S_LEN - 1;
      *(h8*)&bT[n * 32 + kv * 8] =
          *(const h8*)(qb + (size_t)gs * DH + k0 + kv * 8);
    }
    __syncthreads();

    h16 a = fragA(&aT[(16 * wave + l15) * 32 + koffA]);
    #pragma unroll
    for (int t = 0; t < 8; ++t) {
      h16 b = fragB(&bT[(16 * t + l15) * 32 + koffB]);
      acc[t] = wmma16(a, b, acc[t]);
    }
  }

  _Float16* rb = relws + (size_t)bh * LPAD * SPAD;
  #pragma unroll
  for (int t = 0; t < 8; ++t) {
    #pragma unroll
    for (int r = 0; r < 8; ++r) {
      int l = m0 + 16 * wave + r + koffA;
      int s = n0 + 16 * t + l15;
      if (l < LREL && s < S_LEN)
        rb[(size_t)l * SPAD + s] = (_Float16)acc[t][r];
    }
  }
}

// ---------------------------------------------------------------------------
// Kernel 3: flash attention with rel bias.
// Block = 256 thr = 8 waves. Waves (2p, 2p+1) share q-tile p (16 queries),
// each owns half the 384 output dims (12 x 16-wide tiles). grid (15, 16).
// ---------------------------------------------------------------------------
__global__ __launch_bounds__(256)
void mctc_attn_kernel(const _Float16* __restrict__ qws,
                      const _Float16* __restrict__ kws,
                      const _Float16* __restrict__ vtws,
                      const _Float16* __restrict__ relws,
                      float* __restrict__ out) {
  __shared__ _Float16 qtile[4][16 * DH];   // 48 KB
  __shared__ _Float16 ptile[8][16 * 32];   //  8 KB (per-wave prob tiles)

  const int tid = threadIdx.x;
  const int bh  = blockIdx.y;

  // stage 4 q-tiles (64 queries) into LDS, f16 -> f16 copy
  #pragma unroll
  for (int i = 0; i < 12; ++i) {
    int f   = tid + 256 * i;                 // < 3072 vectors of 8 halves
    int p   = f / 768;
    int rem = f % 768;
    int row = rem / 48, kv = rem % 48;
    int qr  = blockIdx.x * 64 + p * 16 + row;
    if (qr > S_LEN - 1) qr = S_LEN - 1;
    *(h8*)&qtile[p][row * DH + kv * 8] =
        *(const h8*)(qws + ((size_t)bh * S_LEN + qr) * DH + kv * 8);
  }
  __syncthreads();

  const int wave = tid >> 5, lane = tid & 31;
  const int pair = wave >> 1, half = wave & 1;
  const int q0 = blockIdx.x * 64 + pair * 16;
  if (q0 >= S_LEN) return;                    // uniform per wave; no barriers after

  const int l15 = lane & 15, hsel = lane >> 4;
  const int koffA = hsel * 8, koffB = hsel * 8;

  const _Float16* kb = kws  + (size_t)bh * S_LEN * DH;
  const _Float16* vb = vtws + (size_t)bh * DH * SPAD;
  const _Float16* rb = relws + (size_t)bh * LPAD * SPAD;
  const _Float16* qt = &qtile[pair][0];
  _Float16*       pt = &ptile[wave][0];

  f8 acc[12];
  #pragma unroll
  for (int t = 0; t < 12; ++t) acc[t] = zero8();
  float mr[8], lr[8];
  #pragma unroll
  for (int r = 0; r < 8; ++r) { mr[r] = NEG_INF; lr[r] = 0.f; }

  for (int k0 = 0; k0 < S_LEN; k0 += 32) {
    const int key0 = k0 + l15, key1 = key0 + 16;
    const int kr0 = (key0 > S_LEN - 1) ? (S_LEN - 1) : key0;
    const int kr1 = (key1 > S_LEN - 1) ? (S_LEN - 1) : key1;
    const _Float16* krow0 = kb + (size_t)kr0 * DH;
    const _Float16* krow1 = kb + (size_t)kr1 * DH;

    f8 s0 = zero8(), s1 = zero8();
    #pragma unroll
    for (int kk = 0; kk < DH; kk += 32) {
      h16 a  = fragA(qt + l15 * DH + kk + koffA);
      h16 b0 = fragB(krow0 + kk + koffB);
      h16 b1 = fragB(krow1 + kk + koffB);
      s0 = wmma16(a, b0, s0);
      s1 = wmma16(a, b1, s1);
    }

    // rel bias, key mask, online softmax stats
    float sc[8];
    #pragma unroll
    for (int r = 0; r < 8; ++r) {
      int qq = q0 + r + koffA;                 // C layout row
      int qc = (qq > S_LEN - 1) ? (S_LEN - 1) : qq;
      int l0 = 919 + key0 - qq; l0 = l0 < 0 ? 0 : (l0 > LPAD - 10 ? LPAD - 10 : l0);
      int l1 = 919 + key1 - qq; l1 = l1 < 0 ? 0 : (l1 > LPAD - 10 ? LPAD - 10 : l1);
      float v0 = s0[r] + (float)rb[(size_t)l0 * SPAD + qc];
      float v1 = s1[r] + (float)rb[(size_t)l1 * SPAD + qc];
      if (key0 >= S_LEN) v0 = NEG_INF;
      if (key1 >= S_LEN) v1 = NEG_INF;

      float tm = fmaxf(v0, v1);
      tm = fmaxf(tm, __shfl_xor(tm, 1, 32));
      tm = fmaxf(tm, __shfl_xor(tm, 2, 32));
      tm = fmaxf(tm, __shfl_xor(tm, 4, 32));
      tm = fmaxf(tm, __shfl_xor(tm, 8, 32));
      float mn    = fmaxf(mr[r], tm);
      float scale = __expf(mr[r] - mn);        // exp(-inf)=0 on first tile
      float p0 = __expf(v0 - mn);
      float p1 = __expf(v1 - mn);
      float rs = p0 + p1;
      rs += __shfl_xor(rs, 1, 32);
      rs += __shfl_xor(rs, 2, 32);
      rs += __shfl_xor(rs, 4, 32);
      rs += __shfl_xor(rs, 8, 32);
      lr[r] = lr[r] * scale + rs;
      mr[r] = mn;
      sc[r] = scale;
      pt[(r + koffA) * 32 + l15]      = (_Float16)p0;
      pt[(r + koffA) * 32 + l15 + 16] = (_Float16)p1;
    }
    #pragma unroll
    for (int t = 0; t < 12; ++t) {
      #pragma unroll
      for (int r = 0; r < 8; ++r) acc[t][r] *= sc[r];
    }

    // wave-local LDS RAW: make prob stores visible before fragment reload
    asm volatile("s_wait_dscnt 0" ::: "memory");
    h16 ap = fragA(pt + l15 * 32 + koffA);

    #pragma unroll
    for (int t = 0; t < 12; ++t) {
      int dcol = half * 192 + 16 * t + l15;
      h16 bv = fragB(vb + (size_t)dcol * SPAD + k0 + koffB);
      acc[t] = wmma16(ap, bv, acc[t]);
    }
  }

  const int b = bh >> 2, h = bh & 3;
  #pragma unroll
  for (int r = 0; r < 8; ++r) {
    int qq = q0 + r + koffA;
    if (qq < S_LEN) {
      float inv = 1.0f / lr[r];
      float* orow = out + ((size_t)(b * S_LEN + qq)) * DMODEL + h * DH + half * 192;
      #pragma unroll
      for (int t = 0; t < 12; ++t) orow[16 * t + l15] = acc[t][r] * inv;
    }
  }
}

// ---------------------------------------------------------------------------
extern "C" void kernel_launch(void* const* d_in, const int* in_sizes, int n_in,
                              void* d_out, int out_size, void* d_ws, size_t ws_size,
                              hipStream_t stream) {
  (void)in_sizes; (void)n_in; (void)out_size; (void)ws_size;
  const float* hid = (const float*)d_in[0];
  const float* qw  = (const float*)d_in[1];
  const float* kw  = (const float*)d_in[2];
  const float* vw  = (const float*)d_in[3];
  const float* emb = (const float*)d_in[4];
  float* out = (float*)d_out;

  char* ws = (char*)d_ws;
  const size_t qBytes  = (size_t)16 * S_LEN * DH * sizeof(_Float16);   // 11.3 MB
  const size_t vtBytes = (size_t)16 * DH * SPAD * sizeof(_Float16);    // 11.4 MB
  _Float16* qws   = (_Float16*)(ws);
  _Float16* kws   = (_Float16*)(ws + qBytes);
  _Float16* vtws  = (_Float16*)(ws + 2 * qBytes);
  _Float16* relws = (_Float16*)(ws + 2 * qBytes + vtBytes);            // 55.1 MB

  dim3 g1((MROWS + 127) / 128, DMODEL / 128, 3);       // (29, 12, 3)
  mctc_qkv_kernel<<<g1, 256, 0, stream>>>(hid, qw, kw, vw, qws, kws, vtws);

  dim3 g2((LREL + 127) / 128, (S_LEN + 127) / 128, 16); // (15, 8, 16)
  mctc_rel_kernel<<<g2, 256, 0, stream>>>(emb, qws, relws);

  dim3 g3((S_LEN + 63) / 64, 16);                       // (15, 16)
  mctc_attn_kernel<<<g3, 256, 0, stream>>>(qws, kws, vtws, relws, out);
}
